// Mama_1984274891157
// MI455X (gfx1250) — matrix operand order repaired and checked by reading, not compiled, not run on hardware
//
#include <hip/hip_runtime.h>
#include <stdint.h>

// ---- problem constants (match reference) ----
#define B_Q      128
#define DIM      512
#define N_CORPUS 500000
#define NBLK     512          // blocks for the streaming scan kernel
#define CT       16           // corpus rows per tile
#define NTILES   (N_CORPUS / CT)   // 31250, exact
#define CSTRIDE  520          // padded ushort row stride in LDS (bank-conflict avoidance)

typedef __attribute__((ext_vector_type(16))) __bf16 v16bf;
typedef __attribute__((ext_vector_type(8)))  float  v8f;
typedef __attribute__((ext_vector_type(4)))  float  f32x4;

union AB { uint4 u[2]; v16bf v; };

__device__ __forceinline__ unsigned short f2bf(float f) {
    unsigned u = __float_as_uint(f);
    u += 0x7FFFu + ((u >> 16) & 1u);      // round-to-nearest-even
    return (unsigned short)(u >> 16);
}

__device__ __forceinline__ void topk3_insert(float (&t)[3], int (&x)[3], float s, int ii) {
    if (s > t[2]) {
        if (s > t[0])      { t[2]=t[1]; x[2]=x[1]; t[1]=t[0]; x[1]=x[0]; t[0]=s; x[0]=ii; }
        else if (s > t[1]) { t[2]=t[1]; x[2]=x[1]; t[1]=s;    x[1]=ii; }
        else               { t[2]=s;    x[2]=ii; }
    }
}

__device__ __forceinline__ void wait_async0() {
#if __has_builtin(__builtin_amdgcn_s_wait_asynccnt)
    __builtin_amdgcn_s_wait_asynccnt(0);
#else
    asm volatile("s_wait_asynccnt 0x0" ::: "memory");
#endif
}

// ---------------- K1: query projection + L2 normalize -> bf16 ----------------
__global__ __launch_bounds__(256) void proj_kernel(const float* __restrict__ qr,
                                                   const float* __restrict__ W,
                                                   unsigned short* __restrict__ qnb) {
    __shared__ float row[DIM];
    __shared__ float red[256];
    const int b = blockIdx.x, tid = threadIdx.x;
    row[tid]       = qr[b * DIM + tid];
    row[tid + 256] = qr[b * DIM + tid + 256];
    __syncthreads();
    float a0 = 0.f, a1 = 0.f;
    for (int k = 0; k < DIM; ++k) {
        const float r = row[k];
        a0 = fmaf(r, W[k * DIM + tid],       a0);
        a1 = fmaf(r, W[k * DIM + tid + 256], a1);
    }
    red[tid] = a0 * a0 + a1 * a1;
    __syncthreads();
    for (int s = 128; s > 0; s >>= 1) {
        if (tid < s) red[tid] += red[tid + s];
        __syncthreads();
    }
    const float rinv = 1.0f / fmaxf(sqrtf(red[0]), 1e-6f);
    qnb[b * DIM + tid]       = f2bf(a0 * rinv);
    qnb[b * DIM + tid + 256] = f2bf(a1 * rinv);
}

// ---- async DMA: 16x512 f32 corpus tile, global -> LDS staging (no VGPR data) ----
__device__ __forceinline__ void async_load_tile(const float* __restrict__ corpus, int ct, int tid,
                                                const f32x4* fbuf /*LDS*/) {
    const char* src = reinterpret_cast<const char*>(corpus + (size_t)ct * CT * DIM);
    #pragma unroll
    for (int t = 0; t < 8; ++t) {
        const int i4 = tid + t * 256;                       // 0..2047 x 16B
        const unsigned lds = (unsigned)(size_t)(fbuf + i4); // LDS byte offset = addr[31:0]
        const char* g = src + (size_t)i4 * 16;
        asm volatile("global_load_async_to_lds_b128 %0, %1, off th:TH_LOAD_NT"
                     :: "v"(lds), "v"(g) : "memory");
    }
}

// ---- conversion pass: LDS f32 staging -> bf16 tile + per-row sum of squares ----
__device__ __forceinline__ void stage_from_fbuf(const f32x4* __restrict__ fbuf, int tid,
                                                unsigned short* __restrict__ cb,
                                                float* __restrict__ sq) {
    #pragma unroll
    for (int t = 0; t < 8; ++t) {
        const int i4 = tid + t * 256;            // 0..2047
        const int e = i4 * 4;
        const int r = e >> 9;
        const int c = e & 511;
        const f32x4 v = fbuf[i4];
        atomicAdd(&sq[r], v.x*v.x + v.y*v.y + v.z*v.z + v.w*v.w);
        uint2 pk;
        pk.x = (unsigned)f2bf(v.x) | ((unsigned)f2bf(v.y) << 16);
        pk.y = (unsigned)f2bf(v.z) | ((unsigned)f2bf(v.w) << 16);
        *reinterpret_cast<uint2*>(&cb[r * CSTRIDE + c]) = pk;
    }
}

// ---- 16x16 score tile: K=512 as 16 x wmma_f32_16x16x32_bf16, B double-buffered ----
__device__ __forceinline__ v8f tile_wmma(const AB (&afrag)[16],
                                         const unsigned short* __restrict__ cb,
                                         int ln, int hi) {
    v8f acc = {0.f,0.f,0.f,0.f,0.f,0.f,0.f,0.f};
    // lane (ln,hi): column n=ln, chunk kk covers K = kk*32 + hi*8 + [0..7] and kk*32 + 16 + hi*8 + [0..7]
    const uint4* base = reinterpret_cast<const uint4*>(&cb[ln * CSTRIDE + hi * 8]);
    AB bf[2];
    bf[0].u[0] = base[0];
    bf[0].u[1] = base[2];
    #pragma unroll
    for (int kk = 0; kk < 16; ++kk) {
        const int nk = kk + 1;
        if (nk < 16) {                       // load-ahead-by-one: hides ds latency behind WMMA
            bf[nk & 1].u[0] = base[nk * 4];
            bf[nk & 1].u[1] = base[nk * 4 + 2];
        }
        acc = __builtin_amdgcn_wmma_f32_16x16x32_bf16(false, afrag[kk].v, false, bf[kk & 1].v,
                                                      (short)0, acc, false, false);
    }
    return acc;
}

// ------- K2: fused corpus-norm + cosine scores (bf16 WMMA) + local top-3 -------
//  async global->LDS prefetch of the next tile overlaps the WMMAs on the current one
__global__ __launch_bounds__(256) void score_topk_kernel(const float* __restrict__ corpus,
                                                         const unsigned short* __restrict__ qnb,
                                                         float* __restrict__ candS,
                                                         int*   __restrict__ candI) {
    __shared__ f32x4 fbuf[CT * DIM / 4];               // 32 KB fp32 async staging
    __shared__ unsigned short cbuf[2][CT * CSTRIDE];   // bf16 corpus tiles, padded rows
    __shared__ float sumsq[2][CT];
    __shared__ float rinvs[2][CT];

    const int tid  = threadIdx.x;
    const int wave = tid >> 5;      // 0..7 -> query tile
    const int lane = tid & 31;
    const int hi   = lane >> 4;     // half-group select
    const int ln   = lane & 15;

    // A fragments: this wave's 16 query rows, K=512 in 16 chunks of 32, VGPR-resident.
    AB afrag[16];
    {
        const int mrow = wave * 16 + ln;
        const uint4* gq = reinterpret_cast<const uint4*>(qnb + (size_t)mrow * DIM);
        #pragma unroll
        for (int kk = 0; kk < 16; ++kk) {
            // lane (ln,hi): K = kk*32 + hi*8 + [0..7]  and  kk*32 + 16 + hi*8 + [0..7]
            afrag[kk].u[0] = gq[kk * 4 + hi];
            afrag[kk].u[1] = gq[kk * 4 + 2 + hi];
        }
    }

    float ts[8][3]; int ti[8][3];
    #pragma unroll
    for (int v = 0; v < 8; ++v) {
        #pragma unroll
        for (int j = 0; j < 3; ++j) { ts[v][j] = -1e30f; ti[v][j] = -1; }
    }

    int ct = blockIdx.x;
    if (ct < NTILES) {
        // ---- prologue: DMA + stage first tile into buffer 0 ----
        if (tid < CT) sumsq[0][tid] = 0.f;
        async_load_tile(corpus, ct, tid, fbuf);
        wait_async0();
        __syncthreads();
        stage_from_fbuf(fbuf, tid, cbuf[0], sumsq[0]);
        __syncthreads();
        if (tid < CT) rinvs[0][tid] = 1.0f / fmaxf(sqrtf(sumsq[0][tid]), 1e-6f);
        __syncthreads();

        int p = 0;
        for (;;) {
            const int  ctn  = ct + NBLK;
            const bool more = ctn < NTILES;

            // DMA the next tile into LDS staging while we run the WMMAs
            if (more) async_load_tile(corpus, ctn, tid, fbuf);
            if (tid < CT) sumsq[p ^ 1][tid] = 0.f;

            // compute current 16x16 score tile per wave (K=512, 16 x bf16 WMMA)
            const v8f   acc  = tile_wmma(afrag, cbuf[p], ln, hi);
            const float rn   = rinvs[p][ln];
            const int   nidx = ct * CT + ln;
            #pragma unroll
            for (int v = 0; v < 8; ++v) topk3_insert(ts[v], ti[v], acc[v] * rn, nidx);

            wait_async0();                   // my wave's DMA into fbuf has landed
            __syncthreads();                 // everyone's DMA + compute done; zeroing visible
            if (!more) break;

            stage_from_fbuf(fbuf, tid, cbuf[p ^ 1], sumsq[p ^ 1]);
            __syncthreads();
            if (tid < CT) rinvs[p ^ 1][tid] = 1.0f / fmaxf(sqrtf(sumsq[p ^ 1][tid]), 1e-6f);
            __syncthreads();

            p ^= 1;
            ct = ctn;
        }
    }

    // butterfly merge of the 16 lanes within each half-group (masks 1,2,4,8)
    #pragma unroll
    for (int mask = 1; mask <= 8; mask <<= 1) {
        #pragma unroll
        for (int v = 0; v < 8; ++v) {
            float ps[3]; int pi[3];
            #pragma unroll
            for (int j = 0; j < 3; ++j) {
                ps[j] = __shfl_xor(ts[v][j], mask, 32);
                pi[j] = __shfl_xor(ti[v][j], mask, 32);
            }
            #pragma unroll
            for (int j = 0; j < 3; ++j) topk3_insert(ts[v], ti[v], ps[j], pi[j]);
        }
    }

    if (ln == 0) {   // lanes 0 and 16 each own 8 query rows
        #pragma unroll
        for (int v = 0; v < 8; ++v) {
            const int qrow = wave * 16 + hi * 8 + v;
            const size_t base = ((size_t)qrow * NBLK + blockIdx.x) * 3;
            #pragma unroll
            for (int j = 0; j < 3; ++j) { candS[base + j] = ts[v][j]; candI[base + j] = ti[v][j]; }
        }
    }
}

// ---------------- K3: per-query merge of NBLK*3 candidates -> top-3 ----------------
__global__ __launch_bounds__(128) void final_topk_kernel(const float* __restrict__ candS,
                                                         const int*   __restrict__ candI,
                                                         float* __restrict__ outS,
                                                         int*   __restrict__ outI) {
    __shared__ float ss[128][3];
    __shared__ int   si[128][3];
    const int q = blockIdx.x, tid = threadIdx.x;
    float t[3] = {-1e30f, -1e30f, -1e30f};
    int   x[3] = {-1, -1, -1};
    const int total = NBLK * 3;
    for (int i = tid; i < total; i += 128) {
        topk3_insert(t, x, candS[(size_t)q * total + i], candI[(size_t)q * total + i]);
    }
    #pragma unroll
    for (int j = 0; j < 3; ++j) { ss[tid][j] = t[j]; si[tid][j] = x[j]; }
    __syncthreads();
    for (int s = 64; s > 0; s >>= 1) {
        if (tid < s) {
            #pragma unroll
            for (int j = 0; j < 3; ++j) topk3_insert(ss[tid], si[tid], ss[tid + s][j], si[tid + s][j]);
        }
        __syncthreads();
    }
    if (tid == 0) {
        #pragma unroll
        for (int j = 0; j < 3; ++j) {
            outS[q * 3 + j] = ss[0][j];
            outI[q * 3 + j] = si[0][j];
        }
    }
}

extern "C" void kernel_launch(void* const* d_in, const int* in_sizes, int n_in,
                              void* d_out, int out_size, void* d_ws, size_t ws_size,
                              hipStream_t stream) {
    (void)in_sizes; (void)n_in; (void)out_size; (void)ws_size;
    const float* qr     = (const float*)d_in[0];   // [128,512]
    const float* corpus = (const float*)d_in[1];   // [500000,512]
    const float* W      = (const float*)d_in[2];   // [512,512]
    // d_in[3] = k (always 3)

    // workspace layout
    unsigned short* qnb = (unsigned short*)d_ws;                             // 128*512*2  = 128 KB
    const size_t qnbBytes  = (size_t)B_Q * DIM * sizeof(unsigned short);
    const size_t candBytes = (size_t)B_Q * NBLK * 3 * sizeof(float);         // 768 KB each
    float* candS = (float*)((char*)d_ws + qnbBytes);
    int*   candI = (int*)  ((char*)d_ws + qnbBytes + candBytes);

    float* outS = (float*)d_out;             // [128,3] scores
    int*   outI = (int*)d_out + B_Q * 3;     // [128,3] indices (int32)

    proj_kernel      <<<B_Q,  256, 0, stream>>>(qr, W, qnb);
    score_topk_kernel<<<NBLK, 256, 0, stream>>>(corpus, qnb, candS, candI);
    final_topk_kernel<<<B_Q,  128, 0, stream>>>(candS, candI, outS, outI);
}
